// Q_Attention_44238163149097
// MI455X (gfx1250) — 8-bit matrix operand order repaired and checked by reading, compiled, not run on hardware
//
#include <hip/hip_runtime.h>
#include <cstddef>


typedef int v8i __attribute__((ext_vector_type(8)));
typedef int v4i_b128 __attribute__((vector_size(16)));

#define WB 4
#define WN 1024
#define WC 768
#define WH 12
#define WHD 64
#define WO3 2304
#define WM (WB * WN)

#define AS_GLOBAL __attribute__((address_space(1)))
#define AS_LDS    __attribute__((address_space(3)))

#if defined(__AMDGCN__) && __has_builtin(__builtin_amdgcn_global_load_async_to_lds_b128)
#define HAVE_ASYNC 1
#else
#define HAVE_ASYNC 0
#endif

#if HAVE_ASYNC
#if __has_builtin(__builtin_amdgcn_s_wait_asynccnt)
#define ASYNC_WAIT() __builtin_amdgcn_s_wait_asynccnt(0)
#else
#define ASYNC_WAIT() asm volatile("s_wait_asynccnt 0x0" ::: "memory")
#endif
#endif

__device__ __forceinline__ v8i wmma_iu8(v8i a, v8i b, v8i c) {
  return __builtin_amdgcn_wmma_i32_16x16x64_iu8(true, a, true, b, c, false, false);
}

__device__ __forceinline__ v8i load_a16x64(const signed char* base, int lda, int lane) {
  int row  = lane & 15;
  int koff = (lane >> 4) << 3;
  const signed char* p = base + (size_t)row * lda + koff;
  v8i a;
  a[0] = *(const int*)(p +  0); a[1] = *(const int*)(p +  4);
  a[2] = *(const int*)(p + 16); a[3] = *(const int*)(p + 20);
  a[4] = *(const int*)(p + 32); a[5] = *(const int*)(p + 36);
  a[6] = *(const int*)(p + 48); a[7] = *(const int*)(p + 52);
  return a;
}

__device__ __forceinline__ v8i load_b64x16(const signed char* base, int ldb, int lane) {
  int n  = lane & 15;
  int kb = (lane >> 4) << 3;
  const signed char* p = base + (size_t)n * ldb + kb;
  v8i b;
  const int* q0 = (const int*)p;
  b[0] = q0[0]; b[1] = q0[1]; b[2] = q0[4]; b[3] = q0[5];
  const int* q1 = (const int*)(p + 32);
  b[4] = q1[0]; b[5] = q1[1]; b[6] = q1[4]; b[7] = q1[5];
  return b;
}

__device__ __forceinline__ void stage_panel(signed char* lds, const signed char* g,
                                            int bytes, int tid) {
#if HAVE_ASYNC
  for (int off = tid * 16; off < bytes; off += 256 * 16)
    __builtin_amdgcn_global_load_async_to_lds_b128(
        (AS_GLOBAL v4i_b128*)(g + off), (AS_LDS v4i_b128*)(lds + off), 0, 0);
  ASYNC_WAIT();
#else
  for (int off = tid * 16; off < bytes; off += 256 * 16)
    *(int4*)(lds + off) = *(const int4*)(g + off);
#endif
}

__device__ __forceinline__ float clip87(float t) {
  return fminf(fmaxf(t, -8.f), 7.f);
}

__device__ __forceinline__ int pack4(const float* s, float inv) {
  unsigned r = 0;
#pragma unroll
  for (int i = 0; i < 4; ++i) {
    int iv = (int)rintf(clip87(s[i] * inv));
    r |= (unsigned)(iv & 0xff) << (8 * i);
  }
  return (int)r;
}

__global__ void k_scalars(const float* qkv_act_alpha, const float* q_alpha,
                          const float* k_alpha, const float* v_alpha,
                          const float* attn_alpha, const float* proj_act_alpha,
                          float* S) {
  if (threadIdx.x == 0) {
    float a = 0.f, p = 0.f;
    for (int i = 0; i < WC; ++i) { a += qkv_act_alpha[i]; p += proj_act_alpha[i]; }
    a /= (float)WC; p /= (float)WC;
    float aq = 0.f, ak = 0.f, av = 0.f, am = 0.f;
    for (int i = 0; i < WH; ++i) {
      aq += q_alpha[i]; ak += k_alpha[i]; av += v_alpha[i]; am += attn_alpha[i];
    }
    aq /= (float)WH; ak /= (float)WH; av /= (float)WH; am /= (float)WH;
    S[0] = a; S[1] = aq; S[2] = ak; S[3] = av; S[4] = am; S[5] = p;
    S[6] = 1.4426950408889634f * 0.125f * aq * ak;
  }
}

__global__ void k_quant_x(const float* x0, const float* S, signed char* xq, int n) {
  int i = blockIdx.x * blockDim.x + threadIdx.x;
  if (i >= n) return;
  float a_in = S[0];
  xq[i] = (signed char)(int)rintf(clip87(x0[i] / a_in));
}

__global__ void k_quant_w(const float* w, const float* alpha, signed char* wq,
                          int rows, int cols) {
  int i = blockIdx.x * blockDim.x + threadIdx.x;
  if (i >= rows * cols) return;
  int o = i / cols;
  wq[i] = (signed char)(int)rintf(clip87(w[i] / alpha[o]));
}

__global__ void k_bias(const float* qkv_b, const float* qkv_alpha, const float* S,
                       float* bias_i) {
  int o = blockIdx.x * blockDim.x + threadIdx.x;
  if (o >= WO3) return;
  bias_i[o] = truncf(qkv_b[o] / S[0] / qkv_alpha[o]);
}

__global__ void __launch_bounds__(256) k_gemm_qkv(const signed char* xq,
                                                  const signed char* wq,
                                                  const float* bias_i, int* outI) {
  __shared__ signed char bpanel[64 * WC];
  int tid = threadIdx.x;
  int lane = tid & 31;
  int wv = tid >> 5;
  int mtg = blockIdx.x / (WO3 / 64);
  int ntg = blockIdx.x % (WO3 / 64);

  stage_panel(bpanel, wq + (size_t)ntg * 64 * WC, 64 * WC, tid);
  __syncthreads();

  const signed char* abase = xq + (size_t)(mtg * 128 + wv * 16) * WC;
  v8i acc[4];
#pragma unroll
  for (int nt = 0; nt < 4; ++nt) acc[nt] = (v8i){0, 0, 0, 0, 0, 0, 0, 0};

  for (int kc = 0; kc < WC / 64; ++kc) {
    v8i a = load_a16x64(abase + kc * 64, WC, lane);
    if (kc + 1 < WC / 64) __builtin_prefetch(abase + (kc + 1) * 64, 0, 3);
#pragma unroll
    for (int nt = 0; nt < 4; ++nt) {
      v8i b = load_b64x16(bpanel + (size_t)nt * 16 * WC + kc * 64, WC, lane);
      acc[nt] = wmma_iu8(a, b, acc[nt]);
    }
  }

  int col = lane & 15, half = lane >> 4;
  int mrow = mtg * 128 + wv * 16 + half * 8;
#pragma unroll
  for (int nt = 0; nt < 4; ++nt) {
    int o = ntg * 64 + nt * 16 + col;
    int bi = (int)bias_i[o];
    int* orow = outI + (size_t)mrow * WO3 + o;
#pragma unroll
    for (int r = 0; r < 8; ++r) orow[(size_t)r * WO3] = acc[nt][r] + bi;
  }
}

__global__ void k_norm(const int* qkvI, const float* qkv_alpha,
                       const float* nqw, const float* nqb,
                       const float* nkw, const float* nkb, const float* S,
                       signed char* q2, signed char* k2, signed char* v2t) {
  int idx = blockIdx.x * blockDim.x + threadIdx.x;
  if (idx >= WB * WH * WN) return;
  int n = idx & (WN - 1);
  int h = (idx >> 10) % WH;
  int b = idx / (WH * WN);
  const int* base = qkvI + (size_t)(b * WN + n) * WO3 + h * WHD;
  float a_in = S[0], aq = S[1], ak = S[2], av = S[3];
  size_t bh = (size_t)(b * WH + h);

  {
    const int* x = base;
    const float* al = qkv_alpha + h * WHD;
    int m = 0, v = 0;
    for (int d = 0; d < WHD; ++d) {
      int xi = (int)((float)x[d] * al[d]);
      int c = 1024 / (d + 1);
      int dd = xi - m;
      m += (dd * c) >> 10;
      v += dd * (xi - m);
    }
    float mu = (float)m, varn = (float)v * (1.f / (float)WHD);
    float den = sqrtf(varn) + 1e-5f;
    signed char* out = q2 + (bh * WN + n) * WHD;
    for (int d = 0; d < WHD; ++d) {
      float xf = (float)x[d] * al[d];
      float q1 = (xf - mu) / den;
      float t = (q1 * nqw[d] + nqb[d]) / aq;
      out[d] = (signed char)(int)rintf(clip87(t));
    }
  }
  {
    const int* x = base + WC;
    const float* al = qkv_alpha + WC + h * WHD;
    int m = 0, v = 0;
    for (int d = 0; d < WHD; ++d) {
      int xi = (int)((float)x[d] * al[d]);
      int c = 1024 / (d + 1);
      int dd = xi - m;
      m += (dd * c) >> 10;
      v += dd * (xi - m);
    }
    float mu = (float)m, varn = (float)v * (1.f / (float)WHD);
    float den = sqrtf(varn) + 1e-5f;
    signed char* out = k2 + (bh * WN + n) * WHD;
    for (int d = 0; d < WHD; ++d) {
      float xf = (float)x[d] * al[d];
      float k1 = (xf - mu) / den;
      float t = (k1 * nkw[d] + nkb[d]) / ak;
      out[d] = (signed char)(int)rintf(clip87(t));
    }
  }
  {
    const int* x = base + 2 * WC;
    const float* al = qkv_alpha + 2 * WC + h * WHD;
    signed char* out = v2t + bh * WHD * WN;
    for (int d = 0; d < WHD; ++d) {
      float t = (float)x[d] * (a_in * al[d]) / av;
      out[(size_t)d * WN + n] = (signed char)(int)rintf(clip87(t));
    }
  }
}

__global__ void __launch_bounds__(32) k_attn(const signed char* q2,
                                             const signed char* k2,
                                             const signed char* v2t,
                                             const float* S, signed char* x1q) {
  __shared__ float sc[16 * WN];
  int lane = threadIdx.x;
  int bh = blockIdx.x >> 6;
  int qt = blockIdx.x & 63;
  int b = bh / WH, h = bh % WH;
  float av = S[3], am = S[4], ap = S[5], c1 = S[6];
  const signed char* qbase = q2 + ((size_t)bh * WN + qt * 16) * WHD;
  const signed char* kbase = k2 + (size_t)bh * WN * WHD;
  const signed char* vbase = v2t + (size_t)bh * WHD * WN;
  int col = lane & 15, half = lane >> 4;

  v8i zero = {0, 0, 0, 0, 0, 0, 0, 0};
  v8i a = load_a16x64(qbase, WHD, lane);
  float part[8];
#pragma unroll
  for (int r = 0; r < 8; ++r) part[r] = 0.f;

  for (int kt = 0; kt < WN / 16; ++kt) {
    v8i bm = load_b64x16(kbase + (size_t)kt * 16 * WHD, WHD, lane);
    v8i qk = wmma_iu8(a, bm, zero);
#pragma unroll
    for (int r = 0; r < 8; ++r) {
      float e = c1 * (float)qk[r];
      float f = floorf(e);
      float val = ldexpf(1.0f + (e - f), (int)f);
      sc[(half * 8 + r) * WN + kt * 16 + col] = val;
      part[r] += val;
    }
  }

  float othr[8];
#pragma unroll
  for (int r = 0; r < 8; ++r) {
    float s = part[r];
    for (int off = 8; off; off >>= 1) s += __shfl_xor(s, off, 16);
    part[r] = s;
    othr[r] = __shfl_xor(s, 16, 32);
  }
  int rsel = col & 7;
  float mysum = ((col >> 3) == half) ? part[rsel] : othr[rsel];
  float inv = 1.0f / (mysum * am);
  __syncthreads();

  int row = col;
  int koffA = half * 8;
  v8i acc0 = zero, acc1 = zero, acc2 = zero, acc3 = zero;
  for (int kt2 = 0; kt2 < WN / 64; ++kt2) {
    const float* srow = sc + (size_t)row * WN + kt2 * 64 + koffA;
    v8i a2;
    a2[0] = pack4(srow +  0, inv); a2[1] = pack4(srow +  4, inv);
    a2[2] = pack4(srow + 16, inv); a2[3] = pack4(srow + 20, inv);
    a2[4] = pack4(srow + 32, inv); a2[5] = pack4(srow + 36, inv);
    a2[6] = pack4(srow + 48, inv); a2[7] = pack4(srow + 52, inv);
    acc0 = wmma_iu8(a2, load_b64x16(vbase + (size_t)( 0) * WN + kt2 * 64, WN, lane), acc0);
    acc1 = wmma_iu8(a2, load_b64x16(vbase + (size_t)(16) * WN + kt2 * 64, WN, lane), acc1);
    acc2 = wmma_iu8(a2, load_b64x16(vbase + (size_t)(32) * WN + kt2 * 64, WN, lane), acc2);
    acc3 = wmma_iu8(a2, load_b64x16(vbase + (size_t)(48) * WN + kt2 * 64, WN, lane), acc3);
  }

  float scale = am * av / ap;
  v8i accs[4] = {acc0, acc1, acc2, acc3};
#pragma unroll
  for (int nt = 0; nt < 4; ++nt) {
#pragma unroll
    for (int r = 0; r < 8; ++r) {
      float t = clip87((float)accs[nt][r] * scale);
      int rowg = qt * 16 + half * 8 + r;
      int colg = h * WHD + nt * 16 + col;
      x1q[(size_t)(b * WN + rowg) * WC + colg] = (signed char)(int)rintf(t);
    }
  }
}

__global__ void __launch_bounds__(256) k_gemm_proj(const signed char* x1q,
                                                   const signed char* wp,
                                                   const float* proj_alpha,
                                                   const float* proj_b,
                                                   const float* S, float* out) {
  __shared__ signed char bpanel[64 * WC];
  int tid = threadIdx.x;
  int lane = tid & 31;
  int wv = tid >> 5;
  int mtg = blockIdx.x / (WC / 64);
  int ntg = blockIdx.x % (WC / 64);

  stage_panel(bpanel, wp + (size_t)ntg * 64 * WC, 64 * WC, tid);
  __syncthreads();

  const signed char* abase = x1q + (size_t)(mtg * 128 + wv * 16) * WC;
  v8i acc[4];
#pragma unroll
  for (int nt = 0; nt < 4; ++nt) acc[nt] = (v8i){0, 0, 0, 0, 0, 0, 0, 0};

  for (int kc = 0; kc < WC / 64; ++kc) {
    v8i a = load_a16x64(abase + kc * 64, WC, lane);
    if (kc + 1 < WC / 64) __builtin_prefetch(abase + (kc + 1) * 64, 0, 3);
#pragma unroll
    for (int nt = 0; nt < 4; ++nt) {
      v8i b = load_b64x16(bpanel + (size_t)nt * 16 * WC + kc * 64, WC, lane);
      acc[nt] = wmma_iu8(a, b, acc[nt]);
    }
  }

  int col = lane & 15, half = lane >> 4;
  int mrow = mtg * 128 + wv * 16 + half * 8;
  float ap = S[5];
#pragma unroll
  for (int nt = 0; nt < 4; ++nt) {
    int o = ntg * 64 + nt * 16 + col;
    float sA = proj_alpha[o] * ap;
    float bO = proj_b[o];
    float* orow = out + (size_t)mrow * WC + o;
#pragma unroll
    for (int r = 0; r < 8; ++r) orow[(size_t)r * WC] = (float)acc[nt][r] * sA + bO;
  }
}

static constexpr size_t OFF_SC   = 0;
static constexpr size_t OFF_XQ   = 256;
static constexpr size_t OFF_WQKV = OFF_XQ   + (size_t)WM * WC;
static constexpr size_t OFF_WP   = OFF_WQKV + (size_t)WO3 * WC;
static constexpr size_t OFF_BIAS = OFF_WP   + (size_t)WC * WC;
static constexpr size_t OFF_QKVI = (OFF_BIAS + (size_t)WO3 * 4 + 255) & ~(size_t)255;
static constexpr size_t OFF_Q2   = OFF_QKVI + (size_t)WM * WO3 * 4;
static constexpr size_t OFF_K2   = OFF_Q2   + (size_t)WB * WH * WN * WHD;
static constexpr size_t OFF_V2T  = OFF_K2   + (size_t)WB * WH * WN * WHD;
static constexpr size_t OFF_X1Q  = OFF_V2T  + (size_t)WB * WH * WN * WHD;

extern "C" void kernel_launch(void* const* d_in, const int* in_sizes, int n_in,
                              void* d_out, int out_size, void* d_ws, size_t ws_size,
                              hipStream_t stream) {
  (void)in_sizes; (void)n_in; (void)out_size; (void)ws_size;
  const float* x0             = (const float*)d_in[0];
  const float* qkv_w          = (const float*)d_in[1];
  const float* qkv_b          = (const float*)d_in[2];
  const float* qkv_alpha      = (const float*)d_in[3];
  const float* qkv_act_alpha  = (const float*)d_in[4];
  const float* proj_w         = (const float*)d_in[5];
  const float* proj_b         = (const float*)d_in[6];
  const float* proj_alpha     = (const float*)d_in[7];
  const float* proj_act_alpha = (const float*)d_in[8];
  const float* norm_q_w       = (const float*)d_in[9];
  const float* norm_q_b       = (const float*)d_in[10];
  const float* norm_k_w       = (const float*)d_in[11];
  const float* norm_k_b       = (const float*)d_in[12];
  const float* q_alpha        = (const float*)d_in[13];
  const float* k_alpha        = (const float*)d_in[14];
  const float* v_alpha        = (const float*)d_in[15];
  const float* attn_alpha     = (const float*)d_in[16];

  char* ws = (char*)d_ws;
  float*       S     = (float*)(ws + OFF_SC);
  signed char* xq    = (signed char*)(ws + OFF_XQ);
  signed char* wqkv  = (signed char*)(ws + OFF_WQKV);
  signed char* wp    = (signed char*)(ws + OFF_WP);
  float*       biasI = (float*)(ws + OFF_BIAS);
  int*         qkvI  = (int*)(ws + OFF_QKVI);
  signed char* q2    = (signed char*)(ws + OFF_Q2);
  signed char* k2    = (signed char*)(ws + OFF_K2);
  signed char* v2t   = (signed char*)(ws + OFF_V2T);
  signed char* x1q   = (signed char*)(ws + OFF_X1Q);
  float*       out   = (float*)d_out;

  k_scalars<<<1, 32, 0, stream>>>(qkv_act_alpha, q_alpha, k_alpha, v_alpha,
                                  attn_alpha, proj_act_alpha, S);

  k_quant_x<<<(WM * WC) / 256, 256, 0, stream>>>(x0, S, xq, WM * WC);

  k_quant_w<<<(WO3 * WC) / 256, 256, 0, stream>>>(qkv_w, qkv_alpha, wqkv, WO3, WC);
  k_quant_w<<<(WC * WC) / 256, 256, 0, stream>>>(proj_w, proj_alpha, wp, WC, WC);
  k_bias<<<(WO3 + 255) / 256, 256, 0, stream>>>(qkv_b, qkv_alpha, S, biasI);

  k_gemm_qkv<<<(WM / 128) * (WO3 / 64), 256, 0, stream>>>(xq, wqkv, biasI, qkvI);

  k_norm<<<(WB * WH * WN) / 256, 256, 0, stream>>>(qkvI, qkv_alpha, norm_q_w, norm_q_b,
                                                   norm_k_w, norm_k_b, S, q2, k2, v2t);

  k_attn<<<WB * WH * (WN / 16), 32, 0, stream>>>(q2, k2, v2t, S, x1q);

  k_gemm_proj<<<(WM / 128) * (WC / 64), 256, 0, stream>>>(x1q, wp, proj_alpha,
                                                          proj_b, S, out);
}
